// DSGraphG_58926951301765
// MI455X (gfx1250) — compile-verified, matrix operand order checked by reading, not run on hardware
//
#include <hip/hip_runtime.h>

// ---------------------------------------------------------------------------
// DSGraphG on MI455X (gfx1250): all dense matmuls via v_wmma_f32_16x16x32_bf16
// (f32 data converted to bf16 with RNE; f32 accumulate), edge message passing
// via wave-per-edge float4 gather + global_atomic_add_f32 scatter (L2 resident:
// gather src + accumulator are 32MB each << 192MB L2).
// ---------------------------------------------------------------------------

typedef __attribute__((ext_vector_type(16))) __bf16        v16bf;
typedef __attribute__((ext_vector_type(8)))  float         v8f;
typedef __attribute__((ext_vector_type(4)))  unsigned int  u32x4;

#define NMETA 64
#define MSUB  1024
#define DHID  128
#define NNODE (NMETA * MSUB)      // 65536
#define TCOLS (MSUB * DHID)       // 131072  (view [64][131072] == [N][128])

// f32 -> bf16, round-to-nearest-even (bit-exact with jax bf16 cast for normals)
__device__ __forceinline__ unsigned short f2bf(float f) {
  unsigned u = __float_as_uint(f);
  u += 0x7FFFu + ((u >> 16) & 1u);
  return (unsigned short)(u >> 16);
}

// Load one 16x16x32 bf16 WMMA operand fragment (A-matrix layout; B uses the
// same layout when its tile is stored transposed, i.e. [N][K] in LDS).
// Per ISA 7.12.2: lane half h = lane>>4 selects K-chunks [kk*32 + 8h, +8) and
// [kk*32 + 16 + 8h, +8): two 16-byte LDS reads.
__device__ __forceinline__ v16bf load_frag(const unsigned short* row, int kk, int hf) {
  union { v16bf v; u32x4 q[2]; } u;
  u.q[0] = *(const u32x4*)(row + kk * 32 + hf * 8);
  u.q[1] = *(const u32x4*)(row + kk * 32 + 16 + hf * 8);
  return u.v;
}

// ---------------------------------------------------------------------------
// C[M x NC] = act( A[M x 128] @ W[128 x NC] + bias ), M multiple of 128.
// Block: 256 threads (8 waves), 128-row tile, full NC in one pass.
// Wave w owns rows [w*16, w*16+16); NC/16 accumulator tiles; K loop = 4.
// ---------------------------------------------------------------------------
template <int NC, bool BIAS, bool RELU>
__global__ void __launch_bounds__(256)
k_mm(const float* __restrict__ A, const float* __restrict__ W,
     const float* __restrict__ bias, float* __restrict__ C) {
  __shared__ unsigned short As[128 * 136];   // A tile, row-major [row][k]
  __shared__ unsigned short Ws[NC * 136];    // W tile, transposed [col][k]
  const int tid  = threadIdx.x;
  const long row0 = (long)blockIdx.x * 128;

  // stage A tile (128x128 f32 -> bf16)
  for (int p = tid; p < 128 * 32; p += 256) {
    int r  = p >> 5;
    int c4 = (p & 31) << 2;
    float4 v = *(const float4*)(A + (row0 + r) * DHID + c4);
    unsigned short* d = As + r * 136 + c4;
    d[0] = f2bf(v.x); d[1] = f2bf(v.y); d[2] = f2bf(v.z); d[3] = f2bf(v.w);
  }
  // stage W transposed (W is [128][NC] row-major -> Ws[col][k])
  for (int p = tid; p < 128 * (NC / 4); p += 256) {
    int k  = p / (NC / 4);
    int c4 = (p % (NC / 4)) << 2;
    float4 v = *(const float4*)(W + k * NC + c4);
    Ws[(c4 + 0) * 136 + k] = f2bf(v.x);
    Ws[(c4 + 1) * 136 + k] = f2bf(v.y);
    Ws[(c4 + 2) * 136 + k] = f2bf(v.z);
    Ws[(c4 + 3) * 136 + k] = f2bf(v.w);
  }
  __syncthreads();

  const int wave = tid >> 5, lane = tid & 31;
  const int hf = lane >> 4, nlo = lane & 15;
  const int mrow0 = wave * 16;
  constexpr int NT = NC / 16;

  v8f acc[NT];
#pragma unroll
  for (int t = 0; t < NT; ++t)
#pragma unroll
    for (int r = 0; r < 8; ++r) acc[t][r] = 0.f;

  const unsigned short* arow = As + (mrow0 + nlo) * 136;
#pragma unroll
  for (int kk = 0; kk < 4; ++kk) {
    v16bf af = load_frag(arow, kk, hf);
#pragma unroll
    for (int t = 0; t < NT; ++t) {
      v16bf bfrag = load_frag(Ws + (t * 16 + nlo) * 136, kk, hf);
      acc[t] = __builtin_amdgcn_wmma_f32_16x16x32_bf16(
          false, af, false, bfrag, (short)0, acc[t], false, false);
    }
  }

#pragma unroll
  for (int t = 0; t < NT; ++t) {
    int col = t * 16 + nlo;
    float b = BIAS ? bias[col] : 0.f;
#pragma unroll
    for (int r = 0; r < 8; ++r) {
      float v = acc[t][r] + b;
      if (RELU) v = fmaxf(v, 0.f);
      C[(row0 + mrow0 + r + hf * 8) * (long)NC + col] = v;
    }
  }
}

// ---------------------------------------------------------------------------
// Meta-mix: Out[64 x 131072] = Am[64 x 64] @ Z[64 x 131072]  (K = 64).
// Block handles all 64 meta rows x 128-column tile; 8 waves: wave w owns
// rows (w&3)*16 and cols (w>>2)*64 (4 accumulator tiles); K loop = 2.
// ---------------------------------------------------------------------------
__global__ void __launch_bounds__(256)
k_mix(const float* __restrict__ Am, const float* __restrict__ Z,
      float* __restrict__ Out) {
  __shared__ unsigned short As[64 * 72];    // Am row-major [n][j]
  __shared__ unsigned short Zs[128 * 72];   // Z tile transposed [col][j]
  const int tid  = threadIdx.x;
  const long col0 = (long)blockIdx.x * 128;

  for (int p = tid; p < 64 * 16; p += 256) {          // 64x64 f32
    int r  = p >> 4;
    int c4 = (p & 15) << 2;
    float4 v = *(const float4*)(Am + r * 64 + c4);
    unsigned short* d = As + r * 72 + c4;
    d[0] = f2bf(v.x); d[1] = f2bf(v.y); d[2] = f2bf(v.z); d[3] = f2bf(v.w);
  }
  for (int p = tid; p < 64 * 32; p += 256) {          // 64 x 128 tile of Z
    int j  = p >> 5;
    int c4 = (p & 31) << 2;
    float4 v = *(const float4*)(Z + (long)j * TCOLS + col0 + c4);
    Zs[(c4 + 0) * 72 + j] = f2bf(v.x);
    Zs[(c4 + 1) * 72 + j] = f2bf(v.y);
    Zs[(c4 + 2) * 72 + j] = f2bf(v.z);
    Zs[(c4 + 3) * 72 + j] = f2bf(v.w);
  }
  __syncthreads();

  const int wave = tid >> 5, lane = tid & 31;
  const int hf = lane >> 4, nlo = lane & 15;
  const int mrow0 = (wave & 3) * 16;
  const int ncol0 = (wave >> 2) * 64;

  v8f acc[4];
#pragma unroll
  for (int t = 0; t < 4; ++t)
#pragma unroll
    for (int r = 0; r < 8; ++r) acc[t][r] = 0.f;

  const unsigned short* arow = As + (mrow0 + nlo) * 72;
#pragma unroll
  for (int kk = 0; kk < 2; ++kk) {
    v16bf af = load_frag(arow, kk, hf);
#pragma unroll
    for (int t = 0; t < 4; ++t) {
      v16bf bfrag = load_frag(Zs + (ncol0 + t * 16 + nlo) * 72, kk, hf);
      acc[t] = __builtin_amdgcn_wmma_f32_16x16x32_bf16(
          false, af, false, bfrag, (short)0, acc[t], false, false);
    }
  }

#pragma unroll
  for (int t = 0; t < 4; ++t)
#pragma unroll
    for (int r = 0; r < 8; ++r)
      Out[(long)(mrow0 + r + hf * 8) * TCOLS + col0 + ncol0 + t * 16 + nlo] =
          acc[t][r];
}

// ---------------------------------------------------------------------------
// Edge message passing: one wave per edge (incl. N self-loops appended).
// msg = Z[src] * dinv[src]*dinv[dst], atomically accumulated into Out[dst].
// Lane l handles floats [4l, 4l+4) of the 128-wide feature row.
// ---------------------------------------------------------------------------
__global__ void __launch_bounds__(256)
k_scatter(const int* __restrict__ src, const int* __restrict__ dst,
          const float* __restrict__ dinv, const float* __restrict__ Z,
          float* __restrict__ Out, int E, int total) {
  int e = blockIdx.x * 8 + (threadIdx.x >> 5);
  if (e >= total) return;
  int lane = threadIdx.x & 31;
  int s, d;
  if (e < E) { s = src[e]; d = dst[e]; }
  else       { s = e - E; d = s; }          // self-loop
  float nrm = dinv[s] * dinv[d];
  float4 v = *(const float4*)(Z + (size_t)s * DHID + lane * 4);
  float* o = Out + (size_t)d * DHID + lane * 4;
  atomicAdd(o + 0, v.x * nrm);
  atomicAdd(o + 1, v.y * nrm);
  atomicAdd(o + 2, v.z * nrm);
  atomicAdd(o + 3, v.w * nrm);
}

// ---------------------------------------------------------------------------
// helpers
// ---------------------------------------------------------------------------
__global__ void k_zerof(float* p, int n) {
  int i = blockIdx.x * 256 + threadIdx.x;
  if (i < n) p[i] = 0.f;
}
__global__ void k_zero4(float4* p, int n4) {
  int i = blockIdx.x * 256 + threadIdx.x;
  if (i < n4) p[i] = make_float4(0.f, 0.f, 0.f, 0.f);
}
__global__ void k_degree(const int* __restrict__ dst, float* __restrict__ deg, int E) {
  int e = blockIdx.x * 256 + threadIdx.x;
  if (e < E) atomicAdd(&deg[dst[e]], 1.0f);
}
__global__ void k_dinv(float* deg, int n) {
  int i = blockIdx.x * 256 + threadIdx.x;
  if (i < n) deg[i] = rsqrtf(deg[i] + 1.0f);   // +1 = self-loop; always > 0
}
// tiny 64x64x64 f32 matmul for adj powers (262 KFLOP: not worth WMMA setup)
__global__ void k_mm64(const float* __restrict__ A, const float* __restrict__ B,
                       float* __restrict__ C) {
  int i = blockIdx.x * 256 + threadIdx.x;     // 4096 outputs
  int r = i >> 6, c = i & 63;
  float s = 0.f;
#pragma unroll 8
  for (int k = 0; k < 64; ++k) s += A[r * 64 + k] * B[k * 64 + c];
  C[i] = s;
}
// x_next = relu(out_acc + sum_i gcn_b[li,i,:])  (bias of all 4 branches)
__global__ void k_relu_bias(const float* __restrict__ acc,
                            const float* __restrict__ b4,  // gcn_b + li*4*128
                            float* __restrict__ xout, int n) {
  int i = blockIdx.x * 256 + threadIdx.x;
  if (i >= n) return;
  int c = i & 127;
  float bs = b4[c] + b4[128 + c] + b4[256 + c] + b4[384 + c];
  xout[i] = fmaxf(acc[i] + bs, 0.f);
}

// ---------------------------------------------------------------------------
extern "C" void kernel_launch(void* const* d_in, const int* in_sizes, int n_in,
                              void* d_out, int out_size, void* d_ws, size_t ws_size,
                              hipStream_t stream) {
  const float* x       = (const float*)d_in[0];
  const int*   sub_adj = (const int*)  d_in[1];
  const float* adj     = (const float*)d_in[2];
  const float* gcn_W   = (const float*)d_in[3];
  const float* gcn_b   = (const float*)d_in[4];
  const float* lin_W1  = (const float*)d_in[5];
  const float* lin_b1  = (const float*)d_in[6];
  const float* lin_W2  = (const float*)d_in[7];
  const float* lin_b2  = (const float*)d_in[8];

  const int N = NNODE;
  const int E = in_sizes[1] / 2;
  const int total = E + N;

  // workspace layout (floats): dinv | A2 | A3 | xcur | outacc | bufA | bufB
  float* ws     = (float*)d_ws;
  float* dinv   = ws;
  float* A2     = dinv + N;
  float* A3     = A2 + 64 * 64;
  float* xcur   = A3 + 64 * 64;
  float* outacc = xcur   + (size_t)N * DHID;
  float* bufA   = outacc + (size_t)N * DHID;
  float* bufB   = bufA   + (size_t)N * DHID;

  const int* srcp = sub_adj;
  const int* dstp = sub_adj + E;

  // --- normalization: deg -> dinv -------------------------------------------
  k_zerof <<<(N + 255) / 256, 256, 0, stream>>>(dinv, N);
  k_degree<<<(E + 255) / 256, 256, 0, stream>>>(dstp, dinv, E);
  k_dinv  <<<(N + 255) / 256, 256, 0, stream>>>(dinv, N);

  // --- cached adjacency powers: adj, adj^2, adj^3 ---------------------------
  k_mm64<<<16, 256, 0, stream>>>(adj, adj, A2);
  k_mm64<<<16, 256, 0, stream>>>(A2, adj, A3);
  const float* Apow[3] = {adj, A2, A3};

  const int mmBlocks  = N / 128;              // 512
  const int mixBlocks = TCOLS / 128;          // 1024
  const int scBlocks  = (total + 7) / 8;

  // --- 3 GCN layers ---------------------------------------------------------
  const float* cur = x;
  for (int li = 0; li < 3; ++li) {
    k_zero4<<<(N * 32 + 255) / 256, 256, 0, stream>>>((float4*)outacc, N * 32);
    for (int br = 0; br < 4; ++br) {
      const float* W = gcn_W + ((size_t)li * 4 + br) * DHID * DHID;
      // z = cur @ W   (WMMA)
      k_mm<128, false, false><<<mmBlocks, 256, 0, stream>>>(cur, W, nullptr, bufA);
      const float* gsrc = bufA;
      if (br > 0) {
        // (A_i x) W == A_i (x W): meta-mix the matmul result (WMMA)
        k_mix<<<mixBlocks, 256, 0, stream>>>(Apow[br - 1], bufA, bufB);
        gsrc = bufB;
      }
      k_scatter<<<scBlocks, 256, 0, stream>>>(srcp, dstp, dinv, gsrc, outacc,
                                              E, total);
    }
    k_relu_bias<<<(N * DHID + 255) / 256, 256, 0, stream>>>(
        outacc, gcn_b + (size_t)li * 4 * DHID, xcur, N * DHID);
    cur = xcur;
  }

  // --- final MLP ------------------------------------------------------------
  k_mm<128, true, true ><<<mmBlocks, 256, 0, stream>>>(cur,  lin_W1, lin_b1, bufA);
  k_mm< 64, true, false><<<mmBlocks, 256, 0, stream>>>(bufA, lin_W2, lin_b2,
                                                       (float*)d_out);
}
